// UDFCdis_816043786553
// MI455X (gfx1250) — matrix operand order repaired and checked by reading, not compiled, not run on hardware
//
#include <hip/hip_runtime.h>

// ---------------- types ----------------
typedef __attribute__((ext_vector_type(16))) __bf16 bf16x16;
typedef __attribute__((ext_vector_type(8)))  float  floatx8;

#define N_ROWS 32768
#define K_CEN  1000
#define K_PAD  1024
#define DDIM   1024
#define APAD   8
#define AST    (DDIM + APAD)   // 1032 halves per LDS row (2064B stride, 16B aligned)

// dynamic LDS: aHi[32][AST] + aLo[32][AST] (bf16 bits) + hsq[32] + red[32][8]
static constexpr size_t SMEM_BYTES =
    (size_t)2 * 32 * AST * 2 + 32 * 4 + 32 * 8 * 4;   // 133,248 B

// ---------------- bf16 split helpers (bit-level, RNE) ----------------
__device__ __forceinline__ unsigned short f32_to_bf16_rne(float f) {
  unsigned u = __builtin_bit_cast(unsigned, f);
  u += 0x7FFFu + ((u >> 16) & 1u);
  return (unsigned short)(u >> 16);
}
__device__ __forceinline__ float bf16_bits_to_f32(unsigned short h) {
  unsigned u = ((unsigned)h) << 16;
  return __builtin_bit_cast(float, u);
}
__device__ __forceinline__ void split2(float x, unsigned short& hi, unsigned short& lo) {
  hi = f32_to_bf16_rne(x);
  lo = f32_to_bf16_rne(x - bf16_bits_to_f32(hi));
}

union BFrag {            // one WMMA 16-bit operand fragment (32 B / lane)
  bf16x16 v;
  uint4   q[2];
};

// ---------------- kernel 1: centroid squared norms (padded) ----------------
__global__ __launch_bounds__(256) void csq_kernel(const float* __restrict__ cen,
                                                  float* __restrict__ csq) {
  int n = blockIdx.x;            // 0..K_PAD-1
  int tid = threadIdx.x;
  float s = 0.f;
  if (n < K_CEN) {
    const float4* p = (const float4*)(cen + (size_t)n * DDIM);
    float4 v = p[tid];           // 256 threads * 4 = 1024
    s = v.x * v.x + v.y * v.y + v.z * v.z + v.w * v.w;
  }
  #pragma unroll
  for (int m = 1; m < 32; m <<= 1) s += __shfl_xor(s, m, 32);
  __shared__ float red[8];
  if ((tid & 31) == 0) red[tid >> 5] = s;
  __syncthreads();
  if (tid == 0) {
    float t = 0.f;
    #pragma unroll
    for (int i = 0; i < 8; ++i) t += red[i];
    csq[n] = t;
  }
}

// ---------------- kernel 2: build pre-swizzled bf16 hi/lo B fragments ----------------
// B-fragment layout as used here with V_WMMA_F32_16X16X32_BF16 (contiguous k, same as A, not ISA table order):
//   lane L: column n = ctBase + (L&15); K values koff..koff+15 (koff = (L>=16)?16:0),
//   stored as 16 consecutive halves (V0..V7, 2 per dword).
// Buffer layout: 32-byte block per (ct, dstep, lane), fully coalesced in main loop.
__global__ __launch_bounds__(256) void prepb_kernel(const float* __restrict__ cen,
                                                    uint4* __restrict__ bhi,
                                                    uint4* __restrict__ blo) {
  int gid  = blockIdx.x * 256 + threadIdx.x;   // 0..65535 = (ct*32+ds)*32+lane
  int lane = gid & 31;
  int ds   = (gid >> 5) & 31;
  int ct   = gid >> 10;                        // 0..63
  int n    = ct * 16 + (lane & 15);
  int koff = ds * 32 + ((lane >> 4) ? 16 : 0);

  union { unsigned short s[16]; uint4 q[2]; } uh, ul;
  if (n < K_CEN) {
    const float* src = cen + (size_t)n * DDIM + koff;
    #pragma unroll
    for (int i = 0; i < 16; ++i) split2(src[i], uh.s[i], ul.s[i]);
  } else {
    #pragma unroll
    for (int i = 0; i < 16; ++i) { uh.s[i] = 0; ul.s[i] = 0; }
  }
  size_t base = (size_t)gid * 2;
  bhi[base + 0] = uh.q[0]; bhi[base + 1] = uh.q[1];
  blo[base + 0] = ul.q[0]; blo[base + 1] = ul.q[1];
}

// ---------------- kernel 3: fused split-bf16 GEMM + Student-t + row-normalize ----------------
__global__ __launch_bounds__(256) void dec_kernel(const float* __restrict__ h,
                                                  const uint4* __restrict__ bhi,
                                                  const uint4* __restrict__ blo,
                                                  const float* __restrict__ csq,
                                                  float* __restrict__ out) {
  extern __shared__ char smem[];
  unsigned short* aHi = (unsigned short*)smem;           // 32 x AST halves
  unsigned short* aLo = aHi + 32 * AST;
  float* hsq = (float*)(aLo + 32 * AST);                 // 32
  float* red = hsq + 32;                                 // 32 rows x 8 waves

  const int tid   = threadIdx.x;
  const int wave  = tid >> 5;
  const int lane  = tid & 31;
  const int laneN = lane & 15;
  const int hi8   = (lane >> 4) * 8;
  const size_t rowBase = (size_t)blockIdx.x * 32;

  if (tid < 32) hsq[tid] = 0.f;
  __syncthreads();

  // ---- stage A block (32 rows x 1024) -> bf16 hi/lo in LDS, fused ||h||^2 ----
  const float4* hp = (const float4*)(h + rowBase * (size_t)DDIM);
  for (int i = 0; i < 32; ++i) {                         // i == row, tid == float4 within row
    float4 v = hp[(size_t)i * 256 + tid];
    float sq = v.x * v.x + v.y * v.y + v.z * v.z + v.w * v.w;
    union { unsigned short s[4]; uint2 u; } ph, pl;
    split2(v.x, ph.s[0], pl.s[0]);
    split2(v.y, ph.s[1], pl.s[1]);
    split2(v.z, ph.s[2], pl.s[2]);
    split2(v.w, ph.s[3], pl.s[3]);
    *(uint2*)(aHi + i * AST + tid * 4) = ph.u;
    *(uint2*)(aLo + i * AST + tid * 4) = pl.u;
    #pragma unroll
    for (int m = 1; m < 32; m <<= 1) sq += __shfl_xor(sq, m, 32);
    if (lane == 0) atomicAdd(&hsq[i], sq);               // ds_add_f32
  }
  __syncthreads();

  // ---- split-bf16 WMMA GEMM: 2 row-tiles x 8 col-tiles per wave ----
  floatx8 acc[2][8];
  {
    floatx8 z = {};
    #pragma unroll
    for (int rt = 0; rt < 2; ++rt)
      #pragma unroll
      for (int j = 0; j < 8; ++j) acc[rt][j] = z;
  }
  const int kbA    = (lane >> 4) * 16;                   // A fragment K sub-offset
  const int ctBase = wave * 8;

  for (int ds = 0; ds < 32; ++ds) {                      // K-dim steps of 32
    const int d0 = ds * 32;
    BFrag aH[2], aL[2];
    #pragma unroll
    for (int rt = 0; rt < 2; ++rt) {
      const unsigned short* pa = aHi + (rt * 16 + laneN) * AST + d0 + kbA;
      aH[rt].q[0] = *(const uint4*)pa;
      aH[rt].q[1] = *(const uint4*)(pa + 8);
      const unsigned short* pl = aLo + (rt * 16 + laneN) * AST + d0 + kbA;
      aL[rt].q[0] = *(const uint4*)pl;
      aL[rt].q[1] = *(const uint4*)(pl + 8);
    }
    #pragma unroll
    for (int j = 0; j < 8; ++j) {
      const int ct = ctBase + j;
      const size_t bidx = (size_t)(((ct * 32 + ds) * 32 + lane) * 2);
      BFrag bH, bL;
      bH.q[0] = bhi[bidx]; bH.q[1] = bhi[bidx + 1];
      bL.q[0] = blo[bidx]; bL.q[1] = blo[bidx + 1];
      #pragma unroll
      for (int rt = 0; rt < 2; ++rt) {
        acc[rt][j] = __builtin_amdgcn_wmma_f32_16x16x32_bf16(
            false, aH[rt].v, false, bH.v, (short)0, acc[rt][j], false, false);
        acc[rt][j] = __builtin_amdgcn_wmma_f32_16x16x32_bf16(
            false, aH[rt].v, false, bL.v, (short)0, acc[rt][j], false, false);
        acc[rt][j] = __builtin_amdgcn_wmma_f32_16x16x32_bf16(
            false, aL[rt].v, false, bH.v, (short)0, acc[rt][j], false, false);
      }
    }
  }

  // ---- epilogue: d2 -> q = 1/(1+d2), masked row sums ----
  float rs[2][8];
  #pragma unroll
  for (int rt = 0; rt < 2; ++rt)
    #pragma unroll
    for (int v = 0; v < 8; ++v) rs[rt][v] = 0.f;

  #pragma unroll
  for (int j = 0; j < 8; ++j) {
    const int col = (ctBase + j) * 16 + laneN;
    const float cs = csq[col];
    const float vmask = (col < K_CEN) ? 1.f : 0.f;
    #pragma unroll
    for (int rt = 0; rt < 2; ++rt) {
      #pragma unroll
      for (int v = 0; v < 8; ++v) {
        float hs = hsq[rt * 16 + hi8 + v];
        float d2 = hs + cs - 2.f * acc[rt][j][v];
        d2 = fmaxf(d2, 0.f);
        float q = 1.0f / (1.0f + d2);
        acc[rt][j][v] = q;
        rs[rt][v] += q * vmask;
      }
    }
  }
  // reduce over the 16 lanes of each row group (rows v+hi8 live in one 16-lane half)
  #pragma unroll
  for (int rt = 0; rt < 2; ++rt)
    #pragma unroll
    for (int v = 0; v < 8; ++v) {
      float s = rs[rt][v];
      s += __shfl_xor(s, 1, 32);
      s += __shfl_xor(s, 2, 32);
      s += __shfl_xor(s, 4, 32);
      s += __shfl_xor(s, 8, 32);
      rs[rt][v] = s;
    }
  if (laneN == 0) {
    #pragma unroll
    for (int rt = 0; rt < 2; ++rt)
      #pragma unroll
      for (int v = 0; v < 8; ++v)
        red[(rt * 16 + hi8 + v) * 8 + wave] = rs[rt][v];
  }
  __syncthreads();

  float inv[2][8];
  #pragma unroll
  for (int rt = 0; rt < 2; ++rt)
    #pragma unroll
    for (int v = 0; v < 8; ++v) {
      const int row = rt * 16 + hi8 + v;
      float t = 0.f;
      #pragma unroll
      for (int w = 0; w < 8; ++w) t += red[row * 8 + w];
      inv[rt][v] = 1.0f / t;
    }

  // ---- normalized coalesced stores (ragged K=1000) ----
  #pragma unroll
  for (int j = 0; j < 8; ++j) {
    const int col = (ctBase + j) * 16 + laneN;
    if (col < K_CEN) {
      #pragma unroll
      for (int rt = 0; rt < 2; ++rt)
        #pragma unroll
        for (int v = 0; v < 8; ++v) {
          const size_t row = rowBase + rt * 16 + hi8 + v;
          out[row * (size_t)K_CEN + col] = acc[rt][j][v] * inv[rt][v];
        }
    }
  }
}

// ---------------- host launcher ----------------
extern "C" void kernel_launch(void* const* d_in, const int* in_sizes, int n_in,
                              void* d_out, int out_size, void* d_ws, size_t ws_size,
                              hipStream_t stream) {
  (void)in_sizes; (void)n_in; (void)out_size; (void)ws_size;
  const float* h   = (const float*)d_in[0];
  const float* cen = (const float*)d_in[1];
  float* out = (float*)d_out;

  char* ws = (char*)d_ws;
  uint4* bhi = (uint4*)ws;                           // 2 MB
  uint4* blo = (uint4*)(ws + (size_t)(2u << 20));    // 2 MB
  float* csq = (float*)(ws + (size_t)(4u << 20));    // 4 KB

  hipFuncSetAttribute((const void*)dec_kernel,
                      hipFuncAttributeMaxDynamicSharedMemorySize, (int)SMEM_BYTES);

  csq_kernel<<<K_PAD, 256, 0, stream>>>(cen, csq);
  prepb_kernel<<<256, 256, 0, stream>>>(cen, bhi, blo);
  dec_kernel<<<N_ROWS / 32, 256, SMEM_BYTES, stream>>>(h, bhi, blo, csq, out);
}
